// VectorQuantizer_25744033972332
// MI455X (gfx1250) — matrix operand order repaired and checked by reading, not compiled, not run on hardware
//
#include <hip/hip_runtime.h>

typedef __attribute__((ext_vector_type(16))) _Float16 v16h;
typedef __attribute__((ext_vector_type(8)))  float    v8f;
typedef __attribute__((ext_vector_type(4)))  float    v4f;
typedef __attribute__((ext_vector_type(2)))  float    v2f;

#define EMBED_DIM 64
#define N_CODES   1024

// Workspace layout (bytes):
//   [0      , 128K) : embT     f16 [1024][64]  (code-major, WMMA B source)
//   [128K   , 384K) : embTf32  f32 [1024][64]  (code-major, exact gather source)
//   [384K   , 388K) : halfnorm f32 [1024]      (0.5 * ||e_k||^2)

__global__ __launch_bounds__(256)
void vq_prep_kernel(const float* __restrict__ embed,
                    _Float16* __restrict__ embT,
                    float* __restrict__ embTf32,
                    float* __restrict__ halfnorm) {
    const int k = blockIdx.x * blockDim.x + threadIdx.x;
    if (k >= N_CODES) return;
    float acc = 0.0f;
#pragma unroll 8
    for (int d = 0; d < EMBED_DIM; ++d) {
        const float v = embed[d * N_CODES + k];     // coalesced across k
        embTf32[k * EMBED_DIM + d] = v;
        embT[k * EMBED_DIM + d]    = (_Float16)v;
        acc += v * v;
    }
    halfnorm[k] = 0.5f * acc;
}

__global__ __launch_bounds__(256)
void vq_main_kernel(const float* __restrict__ inputs,
                    const _Float16* __restrict__ embT,
                    const float* __restrict__ embTf32,
                    const float* __restrict__ halfnorm,
                    float* __restrict__ out,
                    int nRows) {
    __shared__ float s_hn[N_CODES];
    const int tid = threadIdx.x;
    for (int i = tid; i < N_CODES; i += 256) s_hn[i] = halfnorm[i];
    __syncthreads();

    const int lane    = tid & 31;
    const int wave    = tid >> 5;
    const int rowBase = blockIdx.x * 128 + wave * 16;
    const int lmod    = lane & 15;           // row within tile (A) / code within tile (B)
    const int h       = lane >> 4;           // lane-half selector

    // ---- Build A fragments (16 rows x K=64 -> two 16x32 f16 fragments) ----
    // A 16x32 layout (matches B): lane half h holds K = 16h .. 16h+15 contiguous.
    const float* ap = inputs + (size_t)(rowBase + lmod) * EMBED_DIM;
    const int koff  = h * 16;
    v4f p0 = *(const v4f*)(ap + koff);            // chunk0: K 16h..16h+3
    v4f p1 = *(const v4f*)(ap + koff + 4);        //         K 16h+4..16h+7
    v4f p2 = *(const v4f*)(ap + koff + 8);        //         K 16h+8..
    v4f p3 = *(const v4f*)(ap + koff + 12);
    v4f p4 = *(const v4f*)(ap + 32 + koff);       // chunk1 (+32)
    v4f p5 = *(const v4f*)(ap + 32 + koff + 4);
    v4f p6 = *(const v4f*)(ap + 32 + koff + 8);
    v4f p7 = *(const v4f*)(ap + 32 + koff + 12);
    v16h a0, a1;
#pragma unroll
    for (int i = 0; i < 4; ++i) {
        a0[i]      = (_Float16)p0[i];
        a0[4 + i]  = (_Float16)p1[i];
        a0[8 + i]  = (_Float16)p2[i];
        a0[12 + i] = (_Float16)p3[i];
        a1[i]      = (_Float16)p4[i];
        a1[4 + i]  = (_Float16)p5[i];
        a1[8 + i]  = (_Float16)p6[i];
        a1[12 + i] = (_Float16)p7[i];
    }

    // ---- Scan all 64 code tiles, tracking argmax of (x.e - 0.5||e||^2) ----
    float best[8];
    int   bidx[8];
#pragma unroll
    for (int v = 0; v < 8; ++v) { best[v] = -3.402823466e38f; bidx[v] = 0; }

    // B 32x16 f16 layout: lane n (0-15) holds K = 16h .. 16h+15 contiguous.
    const _Float16* bBase = embT + (size_t)lmod * EMBED_DIM + h * 16;

#pragma unroll 2
    for (int t = 0; t < N_CODES / 16; ++t) {
        const _Float16* bp = bBase + (size_t)t * 16 * EMBED_DIM;
        __builtin_prefetch(bp + 16 * EMBED_DIM, 0, 0);       // global_prefetch_b8
        v16h b0 = *(const v16h*)(bp);                        // K-chunk 0
        v16h b1 = *(const v16h*)(bp + 32);                   // K-chunk 1
        v8f acc = {};
        acc = __builtin_amdgcn_wmma_f32_16x16x32_f16(false, a0, false, b0,
                                                     (short)0, acc, false, false);
        acc = __builtin_amdgcn_wmma_f32_16x16x32_f16(false, a1, false, b1,
                                                     (short)0, acc, false, false);
        const float hn   = s_hn[t * 16 + lmod];
        const int   code = t * 16 + lmod;
#pragma unroll
        for (int v = 0; v < 8; ++v) {
            const float s = acc[v] - hn;
            if (s > best[v]) { best[v] = s; bidx[v] = code; }   // strict >: first-occurrence
        }
    }

    // ---- Cross-lane argmax reduction within each 16-lane half ----
    // Tie-break: higher score wins; equal score -> smaller code index.
#pragma unroll
    for (int v = 0; v < 8; ++v) {
        float s = best[v];
        int   i = bidx[v];
#pragma unroll
        for (int m = 1; m < 16; m <<= 1) {
            const float s2 = __shfl_xor(s, m, 32);
            const int   i2 = __shfl_xor(i, m, 32);
            if (s2 > s || (s2 == s && i2 < i)) { s = s2; i = i2; }
        }
        bidx[v] = i;   // all lanes in the half now hold the winner
    }

    // ---- Gather + write outputs (quantized_st | codes | idx-as-float) ----
    const size_t N   = (size_t)nRows;
    float* qst   = out;                 // [N][64]
    float* codes = out + N * 64;        // [N][128]
    float* idxf  = out + N * 192;       // [N]

#pragma unroll
    for (int r = 0; r < 16; ++r) {
        const int k = __shfl(bidx[r & 7], (r < 8) ? 0 : 16, 32);
        const size_t row = (size_t)(rowBase + r);
        const v2f q  = *(const v2f*)(embTf32 + (size_t)k * EMBED_DIM + lane * 2);
        const v2f xi = *(const v2f*)(inputs + row * EMBED_DIM + lane * 2);
        *(v2f*)(qst   + row * 64  + lane * 2)      = q;   // eval-mode ST output == quantized
        *(v2f*)(codes + row * 128 + lane * 2)      = xi;  // concat(inputs, quantized)
        *(v2f*)(codes + row * 128 + 64 + lane * 2) = q;
        if (lane == 0) idxf[row] = (float)k;
    }
}

extern "C" void kernel_launch(void* const* d_in, const int* in_sizes, int n_in,
                              void* d_out, int out_size, void* d_ws, size_t ws_size,
                              hipStream_t stream) {
    const float* inputs = (const float*)d_in[0];   // [32,4096,64] f32
    const float* embed  = (const float*)d_in[1];   // [64,1024]    f32
    float* out = (float*)d_out;

    char* ws = (char*)d_ws;
    _Float16* embT    = (_Float16*)ws;                    // 128 KB
    float*    embTf32 = (float*)(ws + (128 << 10));       // 256 KB
    float*    halfn   = (float*)(ws + (384 << 10));       // 4 KB

    const int nRows = in_sizes[0] / EMBED_DIM;            // 131072

    vq_prep_kernel<<<(N_CODES + 255) / 256, 256, 0, stream>>>(embed, embT, embTf32, halfn);
    vq_main_kernel<<<nRows / 128, 256, 0, stream>>>(inputs, embT, embTf32, halfn, out, nRows);
}
